// HierarchicalGraphProcessor_65635690217532
// MI455X (gfx1250) — compile-verified, matrix operand order checked by reading, not run on hardware
//
#include <hip/hip_runtime.h>

// ---------------------------------------------------------------------------
// CDNA5 (gfx1250) implementation of the 2-level graph-attention network.
// All GEMMs use v_wmma_f32_16x16x32_bf16 (wave32, f32 accumulate).
// Heavy edge term (17.2 GFLOP/level) is fused: le=relu(E@We+be) never leaves
// the WMMA accumulators; it is scaled by eqW and shuffle-reduced to scores.
// This revision: vectorized b128 global staging + packed b64 LDS stores +
// double-buffered (single-barrier) software pipeline in both WMMA kernels.
// ---------------------------------------------------------------------------

typedef __attribute__((ext_vector_type(16))) __bf16 v16bf;
typedef __attribute__((ext_vector_type(8)))  float  v8f;

#define WMMA_BF16(a, b, c) \
  __builtin_amdgcn_wmma_f32_16x16x32_bf16(false, (a), false, (b), (short)0, (c), false, false)

__device__ __forceinline__ unsigned short f2us(float f) {  // fp32 -> bf16 (RNE)
  unsigned u = __builtin_bit_cast(unsigned, f);
  unsigned r = u + 0x7FFFu + ((u >> 16) & 1u);
  return (unsigned short)(r >> 16);
}
__device__ __forceinline__ unsigned pack2(float a, float b) {  // two bf16 in a dword
  return (unsigned)f2us(a) | ((unsigned)f2us(b) << 16);
}

// Load one 16x32 bf16 WMMA operand fragment from LDS.
// rowp = &lds[myrow * stride]; rows indexed by (lane & 15); K distribution per
// CDNA5 ISA 16-bit A layout: k = k0 + ((j<4)?0:16) + 8*(lane>>4) + 2*(j&3) + t.
__device__ __forceinline__ v16bf load_frag(const unsigned short* rowp, int k0, int half) {
  v16bf f;
#pragma unroll
  for (int j = 0; j < 8; ++j) {
    int k = k0 + ((j < 4) ? 0 : 16) + half * 8 + (j & 3) * 2;
    unsigned v = *(const unsigned*)(rowp + k);   // packed pair, 4B aligned
    f[2 * j]     = __builtin_bit_cast(__bf16, (unsigned short)(v & 0xFFFFu));
    f[2 * j + 1] = __builtin_bit_cast(__bf16, (unsigned short)(v >> 16));
  }
  return f;
}

// ---------------------------------------------------------------------------
// Generic fp32-in / fp32-out GEMM via bf16 WMMA, double-buffered staging.
// C[z] = op( A[z](MxK) * B[z](KxN or NxK if transB) + bias )
// Grid: (N/128, M/64, Z). Block: 256 threads (8 waves), wave tile 32x32.
// Requires: lda/ldb multiples of 4, K multiple of 32, 16B-aligned bases.
// ---------------------------------------------------------------------------
__global__ __launch_bounds__(256)
void gemm_bf16_wmma(const float* __restrict__ A, int lda, long sAz,
                    const float* __restrict__ B, int ldb, long sBz, int transB,
                    const float* __restrict__ bias,
                    float* __restrict__ C, int ldc, long sCz,
                    int K, int doRelu) {
  __shared__ unsigned short As[2 * 64 * 36];    // 2 x (64 x 32 bf16, stride 36)
  __shared__ unsigned short Bt[2 * 128 * 36];   // 2 x B^T: 128 (n) x 32 (k)
  A += (long)blockIdx.z * sAz;
  B += (long)blockIdx.z * sBz;
  C += (long)blockIdx.z * sCz;
  const int m0 = blockIdx.y * 64;
  const int n0 = blockIdx.x * 128;
  const int tid = threadIdx.x;
  const int lane = tid & 31, wid = tid >> 5;
  const int half = lane >> 4, l16 = lane & 15;
  const int wm = wid >> 2, wn = wid & 3;        // 2x4 wave grid

  auto stage = [&](int buf, int kk) {
    unsigned short* Ab = As + buf * (64 * 36);
    unsigned short* Bb = Bt + buf * (128 * 36);
    for (int i = tid; i < 64 * 8; i += 256) {   // A: b128 loads, b64 LDS stores
      int r = i >> 3, c4 = (i & 7) * 4;
      const float4 v = *(const float4*)(A + (long)(m0 + r) * lda + kk + c4);
      uint2 p; p.x = pack2(v.x, v.y); p.y = pack2(v.z, v.w);
      *(uint2*)(Ab + r * 36 + c4) = p;
    }
    if (!transB) {                              // transpose through LDS
      for (int i = tid; i < 32 * 32; i += 256) {
        int k = i >> 5, n4 = (i & 31) * 4;
        const float4 v = *(const float4*)(B + (long)(kk + k) * ldb + n0 + n4);
        Bb[(n4 + 0) * 36 + k] = f2us(v.x);
        Bb[(n4 + 1) * 36 + k] = f2us(v.y);
        Bb[(n4 + 2) * 36 + k] = f2us(v.z);
        Bb[(n4 + 3) * 36 + k] = f2us(v.w);
      }
    } else {                                    // already n-major: packed stores
      for (int i = tid; i < 128 * 8; i += 256) {
        int n = i >> 3, c4 = (i & 7) * 4;
        const float4 v = *(const float4*)(B + (long)(n0 + n) * ldb + kk + c4);
        uint2 p; p.x = pack2(v.x, v.y); p.y = pack2(v.z, v.w);
        *(uint2*)(Bb + n * 36 + c4) = p;
      }
    }
  };

  v8f acc[2][2];
#pragma unroll
  for (int i = 0; i < 2; ++i)
#pragma unroll
    for (int j = 0; j < 2; ++j)
#pragma unroll
      for (int r = 0; r < 8; ++r) acc[i][j][r] = 0.0f;

  const int nk = K >> 5;
  stage(0, 0);
  for (int it = 0; it < nk; ++it) {
    __syncthreads();                            // one barrier per K-step
    if (it + 1 < nk) stage((it + 1) & 1, (it + 1) << 5);  // prefetch next tile
    const unsigned short* Ab = As + (it & 1) * (64 * 36);
    const unsigned short* Bb = Bt + (it & 1) * (128 * 36);
    v16bf af0 = load_frag(&Ab[(wm * 32 + l16) * 36], 0, half);
    v16bf af1 = load_frag(&Ab[(wm * 32 + 16 + l16) * 36], 0, half);
    v16bf bf0 = load_frag(&Bb[(wn * 32 + l16) * 36], 0, half);
    v16bf bf1 = load_frag(&Bb[(wn * 32 + 16 + l16) * 36], 0, half);
    acc[0][0] = WMMA_BF16(af0, bf0, acc[0][0]);
    acc[0][1] = WMMA_BF16(af0, bf1, acc[0][1]);
    acc[1][0] = WMMA_BF16(af1, bf0, acc[1][0]);
    acc[1][1] = WMMA_BF16(af1, bf1, acc[1][1]);
  }
#pragma unroll
  for (int im = 0; im < 2; ++im)
#pragma unroll
    for (int jn = 0; jn < 2; ++jn)
#pragma unroll
      for (int r = 0; r < 8; ++r) {
        int m = m0 + wm * 32 + im * 16 + half * 8 + r;  // C/D layout: m=r+8*half
        int n = n0 + wn * 32 + jn * 16 + l16;           //             n=lane&15
        float v = acc[im][jn][r];
        if (bias) v += bias[n];
        if (doRelu) v = fmaxf(v, 0.0f);
        C[(long)m * ldc + n] = v;
      }
}

// ---------------------------------------------------------------------------
// Fused edge-score kernel: one WG per (b, query node nq).
//   s[(2h+b), nq, m] += sum_c relu( (E @ We)[m, h*128+c] + be ) * eqW[h, b*256+nq, c]
// E = edge_fts[b,nq,:,:] (256x128) staged once in LDS as bf16 (66 KB).
// le tiles live only in WMMA accumulators; reduced via wave32 shfl_xor.
// W column-slabs are double-buffered: one barrier per column tile.
// ---------------------------------------------------------------------------
#define EDGE_LDS_BYTES (256 * 132 * 2 + 2 * 16 * 132 * 2 + 512 * 4 + 512 * 4)

__global__ __launch_bounds__(256)
void edge_score_kernel(const float* __restrict__ edgef, const float* __restrict__ We_l,
                       const float* __restrict__ be_l, const float* __restrict__ eqW,
                       float* __restrict__ s) {
  extern __shared__ char smem[];
  unsigned short* E   = (unsigned short*)smem;                           // [256][132]
  unsigned short* Wt0 = (unsigned short*)(smem + 256 * 132 * 2);         // 2 x [16][132]
  float* g            = (float*)(smem + 256 * 132 * 2 + 2 * 16 * 132 * 2); // [512] eqW row
  float* beS          = g + 512;                                           // [512]

  const int b = blockIdx.x >> 8, nqi = blockIdx.x & 255;
  const int tid = threadIdx.x;
  const int lane = tid & 31, wid = tid >> 5;
  const int half = lane >> 4, l16 = lane & 15;

  // Stage E (contiguous 64 KB bf16) with b128 loads + packed b64 LDS stores.
  const float4* Eg4 = (const float4*)(edgef + ((long)(b * 256 + nqi)) * 256 * 128);
  for (int i = tid; i < 8192; i += 256) {       // i = m*32 + c4/4
    int m = i >> 5, c4 = (i & 31) * 4;
    const float4 v = Eg4[i];
    uint2 p; p.x = pack2(v.x, v.y); p.y = pack2(v.z, v.w);
    *(uint2*)(E + m * 132 + c4) = p;
  }
  for (int i = tid; i < 512; i += 256) {
    int h = i >> 7, c = i & 127;
    g[i]   = eqW[((long)h * 512 + b * 256 + nqi) * 128 + c];
    beS[i] = be_l[i];
  }

  // Stage We[:, ct*16 .. +16) transposed into buffer `buf`.
  auto stageWt = [&](int buf, int ct) {
    unsigned short* W = Wt0 + buf * (16 * 132);
    for (int i = tid; i < 512; i += 256) {      // 16*128/4 quads
      int k = i >> 2, n4 = (i & 3) * 4;
      const float4 v = *(const float4*)(We_l + (long)k * 512 + ct * 16 + n4);
      W[(n4 + 0) * 132 + k] = f2us(v.x);
      W[(n4 + 1) * 132 + k] = f2us(v.y);
      W[(n4 + 2) * 132 + k] = f2us(v.z);
      W[(n4 + 3) * 132 + k] = f2us(v.w);
    }
  };

  float redacc[2][8];
  stageWt(0, 0);
  for (int ct = 0; ct < 32; ++ct) {             // 32 column tiles of 16 (8 per head)
    __syncthreads();                            // covers E+g+beS on first pass
    if (ct + 1 < 32) stageWt((ct + 1) & 1, ct + 1);
    const unsigned short* W = Wt0 + (ct & 1) * (16 * 132);
    if ((ct & 7) == 0) {
#pragma unroll
      for (int t = 0; t < 2; ++t)
#pragma unroll
        for (int r = 0; r < 8; ++r) redacc[t][r] = 0.0f;
    }
    const int col = ct * 16 + l16;
    const float gc = g[col], bc = beS[col];
#pragma unroll
    for (int t = 0; t < 2; ++t) {               // two 16-row m-tiles per wave
      const int m_base = (wid + t * 8) * 16;
      v8f acc = {0.f, 0.f, 0.f, 0.f, 0.f, 0.f, 0.f, 0.f};
#pragma unroll
      for (int ks = 0; ks < 4; ++ks) {          // K = 128
        v16bf a  = load_frag(&E[(m_base + l16) * 132], ks * 32, half);
        v16bf bb = load_frag(&W[l16 * 132],            ks * 32, half);
        acc = WMMA_BF16(a, bb, acc);
      }
#pragma unroll
      for (int r = 0; r < 8; ++r) {             // relu + scale, stays in registers
        float v = acc[r] + bc;
        v = v > 0.0f ? v : 0.0f;
        redacc[t][r] += v * gc;
      }
    }
    if ((ct & 7) == 7) {                        // head boundary: reduce 16 columns
      const int h = ct >> 3;
#pragma unroll
      for (int r = 0; r < 8; ++r) {
        float v0 = redacc[0][r], v1 = redacc[1][r];
        for (int off = 1; off < 16; off <<= 1) {
          v0 += __shfl_xor(v0, off, 16);
          v1 += __shfl_xor(v1, off, 16);
        }
        redacc[0][r] = v0;
        redacc[1][r] = v1;
      }
      if (l16 == 0) {
        const long rowOff = (((long)(h * 2 + b)) * 256 + nqi) * 256;
#pragma unroll
        for (int r = 0; r < 8; ++r) {
          int m0 = wid * 16 + half * 8 + r;
          int m1 = (wid + 8) * 16 + half * 8 + r;
          s[rowOff + m0] += redacc[0][r];
          s[rowOff + m1] += redacc[1][r];
        }
      }
    }
  }
}

// ------------------------- small scalar kernels ----------------------------

__global__ void concat_kernel(const float* __restrict__ node, const float* __restrict__ hid,
                              float* __restrict__ x) {
  int t = blockIdx.x * blockDim.x + threadIdx.x;  // 65536
  int c = t & 127, bn = t >> 7;
  x[t] = (c < 64) ? node[bn * 64 + c] : hid[bn * 64 + (c - 64)];
}

__global__ void lg_kernel(const float* __restrict__ graph, const float* __restrict__ Wg_l,
                          const float* __restrict__ bg_l, float* __restrict__ lg) {
  int t = blockIdx.x * blockDim.x + threadIdx.x;  // 1024
  int b = t >> 9, o = t & 511;
  float a = bg_l[o];
  for (int c = 0; c < 128; ++c) a += graph[b * 128 + c] * Wg_l[c * 512 + o];
  lg[t] = fmaxf(a, 0.0f);
}

__global__ void gk_kernel(const float* __restrict__ lg, const float* __restrict__ Wgk_l,
                          const float* __restrict__ bgk_l, float* __restrict__ gk) {
  int t = blockIdx.x * blockDim.x + threadIdx.x;  // 4096
  int b = t >> 11, h = (t >> 9) & 3, d = t & 511;
  float a = bgk_l[d];
  for (int c = 0; c < 128; ++c) a += lg[b * 512 + h * 128 + c] * Wgk_l[c * 512 + d];
  gk[(b * 4 + h) * 512 + d] = a;
}

// srow[(2h+b)*256+n] = eq . bek + gq . gk   (per-query additive score bias)
__global__ void srow_kernel(const float* __restrict__ eq, const float* __restrict__ bek_l,
                            const float* __restrict__ gq, const float* __restrict__ gk,
                            float* __restrict__ srow) {
  int t = blockIdx.x * blockDim.x + threadIdx.x;  // 2048
  int n = t & 255, rem = t >> 8;
  int b = rem & 1, h = rem >> 1;
  const float* eqp = eq + ((long)h * 512 + b * 256 + n) * 512;
  const float* gqp = gq + ((long)h * 512 + b * 256 + n) * 512;
  const float* gkp = gk + (b * 4 + h) * 512;
  float a = 0.0f, c2 = 0.0f;
  for (int d = 0; d < 512; ++d) { a += eqp[d] * bek_l[d]; c2 += gqp[d] * gkp[d]; }
  srow[t] = a + c2;
}

__global__ void softmax_kernel(float* __restrict__ s, const float* __restrict__ srow,
                               const float* __restrict__ adj0) {
  __shared__ float red[256];
  const int row = blockIdx.x;           // (2h+b)*256 + n
  const int n = row & 255;
  const int m = threadIdx.x;
  float v = s[(long)row * 256 + m] + srow[row];
  if (!(adj0[n * 256 + m] > 0.0f)) v = -1e9f;
  red[m] = v; __syncthreads();
  for (int st = 128; st > 0; st >>= 1) { if (m < st) red[m] = fmaxf(red[m], red[m + st]); __syncthreads(); }
  const float mx = red[0]; __syncthreads();
  const float e = __expf(v - mx);
  red[m] = e; __syncthreads();
  for (int st = 128; st > 0; st >>= 1) { if (m < st) red[m] += red[m + st]; __syncthreads(); }
  s[(long)row * 256 + m] = e / red[0];
}

__global__ void repack_v_kernel(const float* __restrict__ V, float* __restrict__ vh) {
  int t = blockIdx.x * blockDim.x + threadIdx.x;  // 262144 : vh[(2h+b)][m][d]
  int d = t & 127, m = (t >> 7) & 255, z = t >> 15;
  int b = z & 1, h = z >> 1;
  vh[t] = V[((long)(b * 256 + m)) * 512 + h * 128 + d];
}

__global__ void combine_kernel(const float* __restrict__ att, float* __restrict__ x) {
  int t = blockIdx.x * blockDim.x + threadIdx.x;  // 65536
  int c = t & 127, n = (t >> 7) & 255, b = t >> 15;
  float mx = -3.4e38f;
  for (int h = 0; h < 4; ++h)
    mx = fmaxf(mx, att[((long)(h * 2 + b) * 256 + n) * 128 + c]);
  x[t] += mx;                                     // skip connection
}

// ---------------------------------------------------------------------------

extern "C" void kernel_launch(void* const* d_in, const int* in_sizes, int n_in,
                              void* d_out, int out_size, void* d_ws, size_t ws_size,
                              hipStream_t stream) {
  (void)in_sizes; (void)n_in; (void)out_size; (void)ws_size;
#define IN(i) ((const float*)d_in[i])
  const float* node   = IN(0);
  const float* edgef  = IN(1);
  const float* graph  = IN(2);
  const float* adj0   = IN(3);   // reference uses adj_mat[0] for every batch
  const float* hidden = IN(4);

  float* ws  = (float*)d_ws;
  float* x   = ws; ws += 65536;        // [512][128]
  float* ln  = ws; ws += 262144;       // [512][512]
  float* Q   = ws; ws += 262144;
  float* Km  = ws; ws += 262144;
  float* V   = ws; ws += 262144;
  float* nq  = ws; ws += 1048576;      // [h][512][512]
  float* nk  = ws; ws += 1048576;
  float* eq  = ws; ws += 1048576;
  float* gq  = ws; ws += 1048576;
  float* eqW = ws; ws += 262144;       // [h][512][128]
  float* lg  = ws; ws += 1024;         // [2][512]
  float* gk  = ws; ws += 4096;         // [b*4+h][512]
  float* srow= ws; ws += 2048;         // [(2h+b)*256+n]
  float* s   = ws; ws += 524288;       // [(2h+b)][256][256]
  float* vh  = ws; ws += 262144;       // [(2h+b)][256][128]
  float* att = ws; ws += 262144;       // [(2h+b)][256][128]

  const dim3 blk(256);
  concat_kernel<<<256, blk, 0, stream>>>(node, hidden, x);

  for (int l = 0; l < 2; ++l) {
    const float *Wn_l  = IN(5)  + l * 65536,  *bn_l  = IN(6)  + l * 512;
    const float *We_l  = IN(7)  + l * 65536,  *be_l  = IN(8)  + l * 512;
    const float *Wg_l  = IN(9)  + l * 65536,  *bg_l  = IN(10) + l * 512;
    const float *Wq_l  = IN(11) + l * 262144, *bq_l  = IN(12) + l * 512;
    const float *Wk_l  = IN(13) + l * 262144, *bk_l  = IN(14) + l * 512;
    const float *Wv_l  = IN(15) + l * 262144, *bv_l  = IN(16) + l * 512;
    const float *Wnq_l = IN(17) + l * 65536,  *bnq_l = IN(18) + l * 512;
    const float *Wnk_l = IN(19) + l * 65536,  *bnk_l = IN(20) + l * 512;
    const float *Weq_l = IN(21) + l * 65536,  *beq_l = IN(22) + l * 512;
    const float *Wek_l = IN(23) + l * 65536,  *bek_l = IN(24) + l * 512;
    const float *Wgq_l = IN(25) + l * 65536,  *bgq_l = IN(26) + l * 512;
    const float *Wgk_l = IN(27) + l * 65536,  *bgk_l = IN(28) + l * 512;

    // ln = relu(x @ Wn + bn)                         M=512 K=128 N=512
    gemm_bf16_wmma<<<dim3(4, 8, 1), blk, 0, stream>>>(x, 128, 0, Wn_l, 512, 0, 0, bn_l, ln, 512, 0, 128, 1);
    // Q/K/V = ln @ W* + b*                           M=512 K=512 N=512
    gemm_bf16_wmma<<<dim3(4, 8, 1), blk, 0, stream>>>(ln, 512, 0, Wq_l, 512, 0, 0, bq_l, Q,  512, 0, 512, 0);
    gemm_bf16_wmma<<<dim3(4, 8, 1), blk, 0, stream>>>(ln, 512, 0, Wk_l, 512, 0, 0, bk_l, Km, 512, 0, 512, 0);
    gemm_bf16_wmma<<<dim3(4, 8, 1), blk, 0, stream>>>(ln, 512, 0, Wv_l, 512, 0, 0, bv_l, V,  512, 0, 512, 0);
    // per-head 128->512 projections (z = head; A offset = 128-col slice of Q/K)
    gemm_bf16_wmma<<<dim3(4, 8, 4), blk, 0, stream>>>(Q,  512, 128, Wnq_l, 512, 0, 0, bnq_l, nq, 512, 262144, 128, 0);
    gemm_bf16_wmma<<<dim3(4, 8, 4), blk, 0, stream>>>(Km, 512, 128, Wnk_l, 512, 0, 0, bnk_l, nk, 512, 262144, 128, 0);
    gemm_bf16_wmma<<<dim3(4, 8, 4), blk, 0, stream>>>(Q,  512, 128, Weq_l, 512, 0, 0, beq_l, eq, 512, 262144, 128, 0);
    gemm_bf16_wmma<<<dim3(4, 8, 4), blk, 0, stream>>>(Q,  512, 128, Wgq_l, 512, 0, 0, bgq_l, gq, 512, 262144, 128, 0);
    // eqW[h] = eq[h] @ Wek^T                         M=512 K=512 N=128
    gemm_bf16_wmma<<<dim3(1, 8, 4), blk, 0, stream>>>(eq, 512, 262144, Wek_l, 512, 0, 1, nullptr, eqW, 128, 65536, 512, 0);
    // graph-level path (tiny)
    lg_kernel<<<4, blk, 0, stream>>>(graph, Wg_l, bg_l, lg);
    gk_kernel<<<16, blk, 0, stream>>>(lg, Wgk_l, bgk_l, gk);
    srow_kernel<<<8, blk, 0, stream>>>(eq, bek_l, gq, gk, srow);
    // s = nq @ nk^T per (b,h); z = 2h+b -> linear 131072*z slice offsets
    gemm_bf16_wmma<<<dim3(2, 4, 8), blk, 0, stream>>>(nq, 512, 131072, nk, 512, 131072, 1, nullptr, s, 256, 65536, 512, 0);
    // fused edge scores: s += sum_c relu(E@We+be) * eqW   (the 17 GFLOP term)
    edge_score_kernel<<<512, blk, EDGE_LDS_BYTES, stream>>>(edgef, We_l, be_l, eqW, s);
    // attention value product
    repack_v_kernel<<<1024, blk, 0, stream>>>(V, vh);
    softmax_kernel<<<2048, blk, 0, stream>>>(s, srow, adj0);
    gemm_bf16_wmma<<<dim3(1, 4, 8), blk, 0, stream>>>(s, 256, 65536, vh, 128, 32768, 0, nullptr, att, 128, 32768, 256, 0);
    combine_kernel<<<256, blk, 0, stream>>>(att, x);  // max over heads + skip
  }
  // out = relu(x @ Wout + bout)                      M=512 K=128 N=512
  gemm_bf16_wmma<<<dim3(4, 8, 1), blk, 0, stream>>>(x, 128, 0, IN(29), 512, 0, 0, IN(30),
                                                    (float*)d_out, 512, 0, 128, 1);
#undef IN
}
